// MultiHeadAttention_27393301413914
// MI455X (gfx1250) — matrix operand order repaired and checked by reading, not compiled, not run on hardware
//
#include <hip/hip_runtime.h>

// ---------------------------------------------------------------------------
// MultiHeadAttention for gfx1250 (MI455X): bf16 WMMA everywhere, single-pass
// fused softmax so the dominant 536 MB attn tensor is written exactly once.
// ---------------------------------------------------------------------------

typedef __attribute__((ext_vector_type(16))) __bf16 v16bf;
typedef __attribute__((ext_vector_type(8)))  __bf16 v8bf;
typedef __attribute__((ext_vector_type(8)))  float  v8f;

#define DMODEL 512
#define NHEAD  8
#define DK     64
#define SEQ    2048
#define BATCH  4
#define ROWS   (BATCH * SEQ)   // 8192

// ---- WMMA helper ----------------------------------------------------------
__device__ __forceinline__ v8f wmma_bf16(v16bf a, v16bf b, v8f c) {
  // v_wmma_f32_16x16x32_bf16 : D = A(16x32) * B(32x16) + C(16x16 f32)
  return __builtin_amdgcn_wmma_f32_16x16x32_bf16(
      /*neg_a=*/false, a, /*neg_b=*/false, b,
      /*c_mod=*/(short)0, c, /*reuse_a=*/false, /*reuse_b=*/false);
}

// A-matrix 16x32 bf16 tile (row-major source, leading dim ld).
// Layout (ISA 7.12.2): lane m (both halves) = row M; lanes 0-15 hold
// K = {0..7, 16..23}, lanes 16-31 hold K = {8..15, 24..31}.
__device__ __forceinline__ v16bf load_a_tile(const __bf16* __restrict__ base,
                                             int ld, int m0, int k0, int lane) {
  int row = lane & 15, hi = lane >> 4;
  const __bf16* p = base + (size_t)(m0 + row) * ld + k0 + hi * 8;
  v8bf a0 = *(const v8bf*)(p);        // K = k0+hi*8   .. +7   (16B)
  v8bf a1 = *(const v8bf*)(p + 16);   // K = k0+16+hi*8 .. +7  (16B)
  v16bf a;
#pragma unroll
  for (int i = 0; i < 8; ++i) { a[i] = a0[i]; a[i + 8] = a1[i]; }
  return a;
}

// B-matrix 32x16 bf16 tile where B[k][n] = W[n][k] (W row-major [N,K], ld=K
// stride). Lane n = column N; lanes 0-15 hold K = {0..7, 16..23}, lanes 16-31
// hold K = {8..15, 24..31} (same order as A) -> 2 16B loads per lane.
__device__ __forceinline__ v16bf load_b_tile(const __bf16* __restrict__ base,
                                             int ld, int n0, int k0, int lane) {
  int row = lane & 15, hi = lane >> 4;
  const __bf16* p = base + (size_t)(n0 + row) * ld + k0 + hi * 8;
  v8bf b0 = *(const v8bf*)(p);
  v8bf b1 = *(const v8bf*)(p + 16);
  v16bf b;
#pragma unroll
  for (int i = 0; i < 8; ++i) { b[i] = b0[i]; b[i + 8] = b1[i]; }
  return b;
}

// ---- elementwise f32 -> bf16 cast (weights) -------------------------------
__global__ void cvt_f32_bf16(const float* __restrict__ src,
                             __bf16* __restrict__ dst, int n) {
  int i = blockIdx.x * blockDim.x + threadIdx.x;
  if (i < n) dst[i] = (__bf16)src[i];
}

// ---- LayerNorm over rows of 512, f32 in -> bf16 out -----------------------
// one block (128 threads = 4 waves) per row
__global__ void ln_rows(const float* __restrict__ x, const float* __restrict__ g,
                        const float* __restrict__ b, __bf16* __restrict__ y) {
  __shared__ float red[4];
  int row = blockIdx.x;
  const float* xr = x + (size_t)row * DMODEL;
  float v[4];
  float s = 0.f;
#pragma unroll
  for (int j = 0; j < 4; ++j) { v[j] = xr[threadIdx.x + 128 * j]; s += v[j]; }
#pragma unroll
  for (int off = 16; off > 0; off >>= 1) s += __shfl_down(s, off, 32);
  if ((threadIdx.x & 31) == 0) red[threadIdx.x >> 5] = s;
  __syncthreads();
  float mu = (red[0] + red[1] + red[2] + red[3]) * (1.0f / DMODEL);
  __syncthreads();
  float q = 0.f;
#pragma unroll
  for (int j = 0; j < 4; ++j) { float d = v[j] - mu; q += d * d; }
#pragma unroll
  for (int off = 16; off > 0; off >>= 1) q += __shfl_down(q, off, 32);
  if ((threadIdx.x & 31) == 0) red[threadIdx.x >> 5] = q;
  __syncthreads();
  float var = (red[0] + red[1] + red[2] + red[3]) * (1.0f / DMODEL);
  float rs = rsqrtf(var + 1e-5f);
#pragma unroll
  for (int j = 0; j < 4; ++j) {
    int c = threadIdx.x + 128 * j;
    y[(size_t)row * DMODEL + c] = (__bf16)((v[j] - mu) * rs * g[c] + b[c]);
  }
}

// ---- GEMM: C[M,N] = A[M,K] @ W[N,K]^T, M=8192, N=K=512 --------------------
// MODE 0: head-split bf16 [B,H,S,64]   (Q / K projections)
// MODE 1: transposed bf16 [B,H,64,S]   (V projection)
// MODE 2: plain f32 [M,N]              (final fc)
// 256 threads = 8 waves; one 16x16 C tile per wave over full K=512.
template <int MODE>
__global__ __launch_bounds__(256) void gemm_xwt(const __bf16* __restrict__ A,
                                                const __bf16* __restrict__ W,
                                                __bf16* __restrict__ dst_bf,
                                                float* __restrict__ dst_f) {
  int lane = threadIdx.x & 31;
  int wave = threadIdx.x >> 5;
  int tile = blockIdx.x * 8 + wave;
  const int TN = DMODEL / 16;  // 32 column tiles
  int m0 = (tile / TN) * 16;
  int n0 = (tile % TN) * 16;
  v8f acc = {};
#pragma unroll 4
  for (int k0 = 0; k0 < DMODEL; k0 += 32) {
    if (k0 + 32 < DMODEL) {  // gfx1250 global_prefetch_b8 hints
      __builtin_prefetch(A + (size_t)(m0 + (lane & 15)) * DMODEL + k0 + 32, 0, 1);
      __builtin_prefetch(W + (size_t)(n0 + (lane & 15)) * DMODEL + k0 + 32, 0, 1);
    }
    v16bf a = load_a_tile(A, DMODEL, m0, k0, lane);
    v16bf b = load_b_tile(W, DMODEL, n0, k0, lane);
    acc = wmma_bf16(a, b, acc);
  }
  int col = lane & 15, hi = lane >> 4;
  int n = n0 + col;
#pragma unroll
  for (int i = 0; i < 8; ++i) {
    int m = m0 + hi * 8 + i;  // C layout: VGPR i -> M=i (lanes 0-15) / M=8+i
    if (MODE == 2) {
      dst_f[(size_t)m * DMODEL + n] = acc[i];
    } else {
      int bb = m >> 11, s = m & 2047;   // batch, seq
      int h = n >> 6, d = n & 63;       // head, dim
      if (MODE == 0)
        dst_bf[(((size_t)(bb * NHEAD + h)) * SEQ + s) * DK + d] = (__bf16)acc[i];
      else
        dst_bf[(((size_t)(bb * NHEAD + h)) * DK + d) * SEQ + s] = (__bf16)acc[i];
    }
  }
}

// ---- fused attention: scores -> softmax -> attn write -> P@V --------------
// One block = one (b,h) x 16 query rows. 16x2048 f32 score panel lives in
// LDS (128 KB; WGP has 320 KB). attn is written to HBM exactly once.
__global__ __launch_bounds__(256) void attention_kernel(
    const __bf16* __restrict__ Qh, const __bf16* __restrict__ Kh,
    const __bf16* __restrict__ Vt, float* __restrict__ attn,
    float* __restrict__ att_out) {
  __shared__ float sS[16][SEQ + 4];  // +4 pad keeps rows 16B-aligned, de-banks
  __shared__ float sO[16][DK];

  int bh = blockIdx.x >> 7;           // (b*8+h), 0..31
  int q0 = (blockIdx.x & 127) * 16;   // query row block
  int lane = threadIdx.x & 31;
  int wave = threadIdx.x >> 5;
  int col = lane & 15, hi = lane >> 4;

  const __bf16* Qb = Qh + ((size_t)bh * SEQ + q0) * DK;
  const __bf16* Kb = Kh + (size_t)bh * SEQ * DK;
  const __bf16* Vb = Vt + (size_t)bh * DK * SEQ;

  for (int i = threadIdx.x; i < 16 * DK; i += 256) ((float*)sO)[i] = 0.f;

  // Phase 1: S = (Q Kᵀ) / 8  via WMMA, panel into LDS.
  v16bf a0 = load_a_tile(Qb, DK, 0, 0, lane);
  v16bf a1 = load_a_tile(Qb, DK, 0, 32, lane);
  for (int j = wave; j < SEQ / 16; j += 8) {
    int n0 = j * 16;
    v16bf b0 = load_b_tile(Kb, DK, n0, 0, lane);   // B[k][n] = K[n][k]
    v16bf b1 = load_b_tile(Kb, DK, n0, 32, lane);
    v8f acc = {};
    acc = wmma_bf16(a0, b0, acc);
    acc = wmma_bf16(a1, b1, acc);
#pragma unroll
    for (int i = 0; i < 8; ++i) sS[hi * 8 + i][n0 + col] = acc[i] * 0.125f;
  }
  __syncthreads();

  // Phase 2: row softmax in LDS. 16 lanes per row (same wave) + shuffles.
  {
    int r = threadIdx.x >> 4;
    int t = threadIdx.x & 15;
    float mx = -3.4e38f;
    for (int c = t; c < SEQ; c += 16) mx = fmaxf(mx, sS[r][c]);
#pragma unroll
    for (int off = 8; off > 0; off >>= 1) mx = fmaxf(mx, __shfl_xor(mx, off, 16));
    float sum = 0.f;
    for (int c = t; c < SEQ; c += 16) {
      float e = __expf(sS[r][c] - mx);
      sS[r][c] = e;
      sum += e;
    }
#pragma unroll
    for (int off = 8; off > 0; off >>= 1) sum += __shfl_xor(sum, off, 16);
    float inv = 1.0f / sum;
    for (int c = t; c < SEQ; c += 16) sS[r][c] *= inv;
  }
  __syncthreads();

  // Phase 3: single coalesced write of normalized attn tile (the 536 MB
  // stream that sets this kernel's runtime floor).
  float* dst = attn + ((size_t)bh * SEQ + q0) * SEQ;
  for (int idx = threadIdx.x; idx < 16 * (SEQ / 4); idx += 256) {
    int r = idx >> 9;        // SEQ/4 = 512 float4 per row
    int c4 = idx & 511;
    float4 vv = *(const float4*)&sS[r][c4 * 4];
    *(float4*)&dst[(size_t)r * SEQ + c4 * 4] = vv;
  }

  // Phase 4: out = P @ V, K=2048 split across waves (4 N-tiles x 2 K-halves),
  // cross-wave reduction via ds_add_f32.
  int nt = wave & 3;
  int kh2 = wave >> 2;
  v8f oacc = {};
  int row = lane & 15;
  for (int k0 = kh2 * (SEQ / 2); k0 < (kh2 + 1) * (SEQ / 2); k0 += 32) {
    v16bf a;
#pragma unroll
    for (int e = 0; e < 8; ++e) {     // A tile straight out of the LDS panel
      a[e]     = (__bf16)sS[row][k0 + hi * 8 + e];
      a[e + 8] = (__bf16)sS[row][k0 + 16 + hi * 8 + e];
    }
    v16bf b = load_b_tile(Vb, SEQ, nt * 16, k0, lane);  // B[k][n] = Vt[n][k]
    oacc = wmma_bf16(a, b, oacc);
  }
#pragma unroll
  for (int i = 0; i < 8; ++i)
    atomicAdd(&sO[hi * 8 + i][nt * 16 + col], oacc[i]);
  __syncthreads();

  // Phase 5: concat-heads layout [B,S,512] f32 for the final LN+fc.
  int b_ = bh >> 3, h_ = bh & 7;
  for (int idx = threadIdx.x; idx < 16 * DK; idx += 256) {
    int r = idx >> 6, d = idx & 63;
    att_out[((size_t)(b_ * SEQ) + q0 + r) * DMODEL + h_ * DK + d] = sO[r][d];
  }
}

// ---------------------------------------------------------------------------
extern "C" void kernel_launch(void* const* d_in, const int* in_sizes, int n_in,
                              void* d_out, int out_size, void* d_ws, size_t ws_size,
                              hipStream_t stream) {
  const float* q   = (const float*)d_in[0];
  const float* k   = (const float*)d_in[1];
  const float* v   = (const float*)d_in[2];
  const float* Wq  = (const float*)d_in[3];
  const float* Wk  = (const float*)d_in[4];
  const float* Wv  = (const float*)d_in[5];
  const float* Wfc = (const float*)d_in[6];
  const float* g   = (const float*)d_in[7];
  const float* bb  = (const float*)d_in[8];

  float* out  = (float*)d_out;                           // [4,2048,512]
  float* attn = out + (size_t)BATCH * SEQ * DMODEL;      // [4,8,2048,2048]

  // Workspace layout (bf16 elements unless noted):
  //   region0: qn, kn, vn          3 * 8192*512 * 2B = 25.2 MB (reused later)
  //   region1: wq, wk, wv, wfc     4 * 512*512 * 2B  =  2.1 MB
  //   region2: Qh, Kh, Vt          3 * 8192*512 * 2B = 25.2 MB
  //   reuse region0: att_out (f32 16.8 MB) + att_n (bf16 8.4 MB)
  char* ws = (char*)d_ws;
  const size_t RSZ = (size_t)ROWS * DMODEL;  // 4,194,304 elements
  __bf16* qn  = (__bf16*)ws;
  __bf16* kn  = qn + RSZ;
  __bf16* vn  = kn + RSZ;
  __bf16* wq  = vn + RSZ;
  __bf16* wk  = wq + DMODEL * DMODEL;
  __bf16* wv  = wk + DMODEL * DMODEL;
  __bf16* wfc = wv + DMODEL * DMODEL;
  __bf16* Qh  = wfc + DMODEL * DMODEL;
  __bf16* Kh  = Qh + RSZ;
  __bf16* Vt  = Kh + RSZ;
  float*  att_out = (float*)ws;                    // overlaps qn+kn (dead then)
  __bf16* att_n   = (__bf16*)(ws + 2 * RSZ * sizeof(__bf16));  // overlaps vn

  // 1) weights -> bf16 (torch [out,in] layout kept: WMMA-B lanes read 16
  //    K each, as two contiguous runs of 8)
  const int nW = DMODEL * DMODEL;
  cvt_f32_bf16<<<(nW + 255) / 256, 256, 0, stream>>>(Wq,  wq,  nW);
  cvt_f32_bf16<<<(nW + 255) / 256, 256, 0, stream>>>(Wk,  wk,  nW);
  cvt_f32_bf16<<<(nW + 255) / 256, 256, 0, stream>>>(Wv,  wv,  nW);
  cvt_f32_bf16<<<(nW + 255) / 256, 256, 0, stream>>>(Wfc, wfc, nW);

  // 2) LayerNorm q,k,v -> bf16
  ln_rows<<<ROWS, 128, 0, stream>>>(q, g, bb, qn);
  ln_rows<<<ROWS, 128, 0, stream>>>(k, g, bb, kn);
  ln_rows<<<ROWS, 128, 0, stream>>>(v, g, bb, vn);

  // 3) projections (WMMA)
  const int gemmBlocks = (ROWS / 16) * (DMODEL / 16) / 8;  // 2048
  gemm_xwt<0><<<gemmBlocks, 256, 0, stream>>>(qn, wq, Qh, nullptr);
  gemm_xwt<0><<<gemmBlocks, 256, 0, stream>>>(kn, wk, Kh, nullptr);
  gemm_xwt<1><<<gemmBlocks, 256, 0, stream>>>(vn, wv, Vt, nullptr);

  // 4) fused attention (scores -> softmax -> attn write -> P@V)
  attention_kernel<<<BATCH * NHEAD * (SEQ / 16), 256, 0, stream>>>(
      Qh, Kh, Vt, attn, att_out);

  // 5) final LN + fc
  ln_rows<<<ROWS, 128, 0, stream>>>(att_out, g, bb, att_n);
  gemm_xwt<2><<<gemmBlocks, 256, 0, stream>>>(att_n, wfc, nullptr, out);
}
